// MyModel_2319282340146
// MI455X (gfx1250) — compile-verified, matrix operand order checked by reading, not run on hardware
//
#include <hip/hip_runtime.h>
#include <hip/hip_bf16.h>
#include <math.h>

typedef __attribute__((ext_vector_type(16))) _Float16 v16h;
typedef __attribute__((ext_vector_type(8)))  float    v8f;
typedef int v4i_ __attribute__((vector_size(16)));   // matches builtin's expected pointee

#define BATCHN 64
#define ICN    200
#define WSN    13
#define HWN    (WSN*WSN)      // 169
#define M1N    (BATCHN*HWN)   // 10816
#define EDN    400
#define NSTN   16
#define DTRN   13
#define NHEADN 8
#define FFN    512
#define EPSF   1e-5f

#if __has_builtin(__builtin_amdgcn_global_load_async_to_lds_b128)
#define HAVE_ASYNC_LDS 1
typedef __attribute__((address_space(1))) v4i_* gv4p;
typedef __attribute__((address_space(3))) v4i_* lv4p;
#endif

__device__ __forceinline__ float sigm_(float x){ return 1.f/(1.f+expf(-x)); }

__device__ __forceinline__ v16h pack16(float4 a, float4 b, float4 c, float4 d){
  v16h r;
  r[0]=(_Float16)a.x;  r[1]=(_Float16)a.y;  r[2]=(_Float16)a.z;  r[3]=(_Float16)a.w;
  r[4]=(_Float16)b.x;  r[5]=(_Float16)b.y;  r[6]=(_Float16)b.z;  r[7]=(_Float16)b.w;
  r[8]=(_Float16)c.x;  r[9]=(_Float16)c.y;  r[10]=(_Float16)c.z; r[11]=(_Float16)c.w;
  r[12]=(_Float16)d.x; r[13]=(_Float16)d.y; r[14]=(_Float16)d.z; r[15]=(_Float16)d.w;
  return r;
}

// ---------------- elementwise ----------------
__global__ void k_copy(float* d, const float* s, long n){
  long i = (long)blockIdx.x*blockDim.x + threadIdx.x; if(i<n) d[i]=s[i];
}
__global__ void k_addip(float* d, const float* a, long n){
  long i = (long)blockIdx.x*blockDim.x + threadIdx.x; if(i<n) d[i]+=a[i];
}

// ---------------- layout transforms ----------------
__global__ void k_nchw2seq(float* seq, const float* x, int B,int C,int Lh){
  long i=(long)blockIdx.x*blockDim.x+threadIdx.x; long tot=(long)B*C*Lh; if(i>=tot) return;
  int l=(int)(i%Lh); int c=(int)((i/Lh)%C); int b=(int)(i/((long)Lh*C));
  seq[((size_t)b*Lh+l)*C + c] = x[i];
}
__global__ void k_seq2nchw(float* x, const float* seq, int B,int C,int Lh){
  long i=(long)blockIdx.x*blockDim.x+threadIdx.x; long tot=(long)B*C*Lh; if(i>=tot) return;
  int l=(int)(i%Lh); int c=(int)((i/Lh)%C); int b=(int)(i/((long)Lh*C));
  x[i] = seq[((size_t)b*Lh+l)*C + c];
}

// ============ WMMA GEMM:  C[M,N] = A[M,K](row,lda) * W[N,K]^T (+bias)(+res)(act) ============
// act: 0 none, 1 relu, 2 softplus.
// 128 threads = 4 waves; each wave owns a 16x32 output tile (2 WMMA per K-step).
// B tile (32 cols x 32 k, f32) staged in LDS via gfx1250 async-to-LDS when available.
// Main K loop assumes lda%4==0 and K%4==0 (true for every call with K>=32 here);
// the K<32 / tail path is scalar + branchless (clamped index + select, no exec branches).
__global__ __launch_bounds__(128) void k_gemm_wmma(
    const float* __restrict__ A, int lda,
    const float* __restrict__ W,
    const float* __restrict__ bias,
    const float* __restrict__ res,
    float* __restrict__ C,
    int M, int K, int N, int act)
{
  __shared__ float Bs[32*32];
  const int tid   = threadIdx.x;
  const int lane  = tid & 31;
  const int wv    = tid >> 5;
  const int n0    = blockIdx.x * 32;
  const int m0    = (blockIdx.y * 4 + wv) * 16;
  const int row   = lane & 15;
  const int half  = lane >> 4;
  const int khalf = half << 3;

  const float* Arow = A + (size_t)min(m0 + row, M - 1) * lda;

  v8f acc0 = {}, acc1 = {};
  const int Kmain = K & ~31;

  for (int k0 = 0; k0 < Kmain; k0 += 32) {
    // ---- stage B tile into LDS: each thread moves two 16B chunks ----
    {
      int flat = tid * 4;
#pragma unroll
      for (int ch = 0; ch < 2; ++ch) {
        int f   = flat + ch * 512;
        int col = f >> 5;
        int kk  = f & 31;
        const float* g = W + (size_t)min(n0 + col, N - 1) * K + k0 + kk;
#if defined(HAVE_ASYNC_LDS)
        __builtin_amdgcn_global_load_async_to_lds_b128((gv4p)g, (lv4p)&Bs[f], 0, 0);
#else
        *(float4*)&Bs[f] = *(const float4*)g;
#endif
      }
#if defined(HAVE_ASYNC_LDS)
      asm volatile("s_wait_asynccnt 0x0" ::: "memory");
#endif
      __syncthreads();
    }
    __builtin_prefetch(Arow + k0 + 64, 0, 1);   // global_prefetch_b8 for next A tile
    // ---- A fragment: two 32B spans per lane (ISA 16-bit A 16x32 layout) ----
    const float4* Ap = (const float4*)(Arow + k0 + khalf);
    v16h av = pack16(Ap[0], Ap[1], Ap[4], Ap[5]);
    // ---- B fragments from LDS (ds_load_b128) ----
    const float4* Bp0 = (const float4*)&Bs[(size_t)row * 32 + khalf];
    const float4* Bp1 = (const float4*)&Bs[(size_t)(row + 16) * 32 + khalf];
    v16h b0 = pack16(Bp0[0], Bp0[1], Bp0[4], Bp0[5]);
    v16h b1 = pack16(Bp1[0], Bp1[1], Bp1[4], Bp1[5]);
    acc0 = __builtin_amdgcn_wmma_f32_16x16x32_f16(false, av, false, b0, (short)0, acc0, false, false);
    acc1 = __builtin_amdgcn_wmma_f32_16x16x32_f16(false, av, false, b1, (short)0, acc1, false, false);
    __syncthreads();
  }

  if (Kmain < K) {              // branchless scalar tail (also the whole K<32 case)
    const int rem = K - Kmain;
    const float* W0 = W + (size_t)min(n0 + row,      N - 1) * K;
    const float* W1 = W + (size_t)min(n0 + 16 + row, N - 1) * K;
    v16h av, b0, b1;
#pragma unroll
    for (int i = 0; i < 16; ++i) {
      int k  = khalf + (i & 7) + ((i >> 3) << 4);
      int kc = Kmain + (k < rem ? k : 0);
      float a  = Arow[kc];
      float w0 = W0[kc];
      float w1 = W1[kc];
      av[i] = (_Float16)(k < rem ? a  : 0.f);
      b0[i] = (_Float16)(k < rem ? w0 : 0.f);
      b1[i] = (_Float16)(k < rem ? w1 : 0.f);
    }
    acc0 = __builtin_amdgcn_wmma_f32_16x16x32_f16(false, av, false, b0, (short)0, acc0, false, false);
    acc1 = __builtin_amdgcn_wmma_f32_16x16x32_f16(false, av, false, b1, (short)0, acc1, false, false);
  }

  // ---- store both 16x16 tiles (ISA f32 C/D layout) ----
#pragma unroll
  for (int t = 0; t < 2; ++t) {
    v8f acc = t ? acc1 : acc0;
    int nc = n0 + t * 16 + row;
    if (nc < N) {
      float bb = bias ? bias[nc] : 0.f;
#pragma unroll
      for (int r = 0; r < 8; ++r) {
        int mr = m0 + r + khalf;
        if (mr < M) {
          float v = acc[r] + bb;
          if (res) v += res[(size_t)mr * N + nc];
          if (act == 1) v = fmaxf(v, 0.f);
          else if (act == 2) v = (v > 20.f) ? v : log1pf(expf(v));
          C[(size_t)mr * N + nc] = v;
        }
      }
    }
  }
}

// ============ WMMA implicit-GEMM conv (NCHW in, OIHW weights, fused bias) ============
template<int KS>
__global__ __launch_bounds__(128) void k_conv_wmma(
    const float* __restrict__ X, const float* __restrict__ Wt,
    const float* __restrict__ bias, float* __restrict__ Y,
    int B, int Cin, int H, int Wd, int Cout, int pad)
{
  const int KSS = KS * KS;
  const int HWl = H * Wd;
  const int M   = B * HWl;
  const int K   = Cin * KSS;
  const int tid   = threadIdx.x;
  const int lane  = tid & 31;
  const int wv    = tid >> 5;
  const int n0    = blockIdx.x * 32;
  const int m0    = (blockIdx.y * 4 + wv) * 16;
  const int row   = lane & 15;
  const int khalf = (lane >> 4) << 3;

  const int mA = min(m0 + row, M - 1);
  const int b  = mA / HWl;
  const int p  = mA % HWl;
  const int oy = p / Wd, ox = p % Wd;
  const float* Xb = X + (size_t)b * Cin * HWl;

  const float* W0 = Wt + (size_t)min(n0 + row,      Cout - 1) * K;
  const float* W1 = Wt + (size_t)min(n0 + 16 + row, Cout - 1) * K;

  v8f acc0 = {}, acc1 = {};
  const int Kmain = K & ~31;

  for (int k0 = 0; k0 < Kmain; k0 += 32) {
    v16h av;
#pragma unroll
    for (int i = 0; i < 16; ++i) {
      int k  = k0 + khalf + (i & 7) + ((i >> 3) << 4);
      int c  = k / KSS;
      int j  = k % KSS;
      int iy = oy + j / KS - pad;
      int ix = ox + j % KS - pad;
      bool inb = (iy >= 0) & (iy < H) & (ix >= 0) & (ix < Wd);
      int iyc = min(max(iy, 0), H - 1);
      int ixc = min(max(ix, 0), Wd - 1);
      float a = Xb[((size_t)c * H + iyc) * Wd + ixc];
      av[i] = (_Float16)(inb ? a : 0.f);
    }
    const float4* Wp0 = (const float4*)(W0 + k0 + khalf);   // K%4==0 for all conv layers
    const float4* Wp1 = (const float4*)(W1 + k0 + khalf);
    v16h b0 = pack16(Wp0[0], Wp0[1], Wp0[4], Wp0[5]);
    v16h b1 = pack16(Wp1[0], Wp1[1], Wp1[4], Wp1[5]);
    acc0 = __builtin_amdgcn_wmma_f32_16x16x32_f16(false, av, false, b0, (short)0, acc0, false, false);
    acc1 = __builtin_amdgcn_wmma_f32_16x16x32_f16(false, av, false, b1, (short)0, acc1, false, false);
  }

  if (Kmain < K) {
    const int rem = K - Kmain;
    v16h av, b0, b1;
#pragma unroll
    for (int i = 0; i < 16; ++i) {
      int k  = khalf + (i & 7) + ((i >> 3) << 4);
      int kc = Kmain + (k < rem ? k : 0);
      int c  = kc / KSS;
      int j  = kc % KSS;
      int iy = oy + j / KS - pad;
      int ix = ox + j % KS - pad;
      bool inb = (k < rem) & (iy >= 0) & (iy < H) & (ix >= 0) & (ix < Wd);
      int iyc = min(max(iy, 0), H - 1);
      int ixc = min(max(ix, 0), Wd - 1);
      float a  = Xb[((size_t)c * H + iyc) * Wd + ixc];
      float w0 = W0[kc];
      float w1 = W1[kc];
      av[i] = (_Float16)(inb ? a : 0.f);
      b0[i] = (_Float16)(k < rem ? w0 : 0.f);
      b1[i] = (_Float16)(k < rem ? w1 : 0.f);
    }
    acc0 = __builtin_amdgcn_wmma_f32_16x16x32_f16(false, av, false, b0, (short)0, acc0, false, false);
    acc1 = __builtin_amdgcn_wmma_f32_16x16x32_f16(false, av, false, b1, (short)0, acc1, false, false);
  }

#pragma unroll
  for (int t = 0; t < 2; ++t) {
    v8f acc = t ? acc1 : acc0;
    int nc = n0 + t * 16 + row;
    if (nc < Cout) {
      float bb = bias ? bias[nc] : 0.f;
#pragma unroll
      for (int r = 0; r < 8; ++r) {
        int mr = m0 + r + khalf;
        if (mr < M) {
          int b2 = mr / HWl, p2 = mr % HWl;
          Y[((size_t)b2 * Cout + nc) * HWl + p2] = acc[r] + bb;
        }
      }
    }
  }
}

// ---------------- BatchNorm (training stats, biased var) ----------------
__global__ void k_bnstats(const float* X, float* st, int B, int C, int HWl){
  int c = blockIdx.x;
  __shared__ float ss[256], s2[256];
  float a = 0.f, b2 = 0.f;
  int tot = B * HWl;
  for (int i = threadIdx.x; i < tot; i += blockDim.x){
    int b = i / HWl, p = i % HWl;
    float v = X[((size_t)b * C + c) * HWl + p];
    a += v; b2 += v * v;
  }
  ss[threadIdx.x] = a; s2[threadIdx.x] = b2; __syncthreads();
  for (int s = blockDim.x >> 1; s > 0; s >>= 1){
    if (threadIdx.x < s){ ss[threadIdx.x]+=ss[threadIdx.x+s]; s2[threadIdx.x]+=s2[threadIdx.x+s]; }
    __syncthreads();
  }
  if (threadIdx.x == 0){
    float m = ss[0] / tot;
    st[c] = m; st[C + c] = s2[0] / tot - m * m;
  }
}
__global__ void k_bnapply(float* X, const float* st, const float* g, const float* bb,
                          int B, int C, int HWl, int relu){
  long i=(long)blockIdx.x*blockDim.x+threadIdx.x; long tot=(long)B*C*HWl; if(i>=tot) return;
  int c = (int)((i / HWl) % C);
  float y = (X[i]-st[c])*rsqrtf(st[C+c]+EPSF)*g[c]+bb[c];
  if (relu) y = fmaxf(y, 0.f);
  X[i] = y;
}

// ---------------- LayerNorm / RMSNorm over rows ----------------
__global__ void k_layernorm(const float* X, const float* g, const float* b, float* Y, int M, int E){
  int r = blockIdx.x; const float* xr = X + (size_t)r * E;
  __shared__ float ss[128], s2[128]; __shared__ float mu, inv;
  float a=0.f, q=0.f;
  for (int i = threadIdx.x; i < E; i += blockDim.x){ float v = xr[i]; a+=v; q+=v*v; }
  ss[threadIdx.x]=a; s2[threadIdx.x]=q; __syncthreads();
  for (int s = blockDim.x >> 1; s > 0; s >>= 1){
    if (threadIdx.x < s){ ss[threadIdx.x]+=ss[threadIdx.x+s]; s2[threadIdx.x]+=s2[threadIdx.x+s]; }
    __syncthreads();
  }
  if (threadIdx.x == 0){ mu = ss[0]/E; inv = rsqrtf(s2[0]/E - mu*mu + EPSF); }
  __syncthreads();
  for (int i = threadIdx.x; i < E; i += blockDim.x)
    Y[(size_t)r*E+i] = (xr[i]-mu)*inv*g[i]+b[i];
}
__global__ void k_rmsnorm(const float* X, const float* w, float* Y, int M, int E){
  int r = blockIdx.x; const float* xr = X + (size_t)r * E;
  __shared__ float s2[128]; __shared__ float inv;
  float q=0.f;
  for (int i = threadIdx.x; i < E; i += blockDim.x){ float v = xr[i]; q+=v*v; }
  s2[threadIdx.x]=q; __syncthreads();
  for (int s = blockDim.x >> 1; s > 0; s >>= 1){
    if (threadIdx.x < s) s2[threadIdx.x]+=s2[threadIdx.x+s];
    __syncthreads();
  }
  if (threadIdx.x == 0) inv = rsqrtf(s2[0]/E + EPSF);
  __syncthreads();
  for (int i = threadIdx.x; i < E; i += blockDim.x)
    Y[(size_t)r*E+i] = xr[i]*inv*w[i];
}

// ---------------- Mamba pieces ----------------
__global__ void k_dwconv_silu(const float* P, const float* w, const float* cb, float* xs,
                              int B, int Ll, int EDl){
  long i=(long)blockIdx.x*blockDim.x+threadIdx.x; long tot=(long)B*Ll*EDl; if(i>=tot) return;
  int e=(int)(i%EDl); int l=(int)((i/EDl)%Ll); int b=(int)(i/((long)EDl*Ll));
  float acc = cb[e];
#pragma unroll
  for (int j = 0; j < 4; ++j){
    int li = l - 3 + j;
    if (li >= 0) acc += w[e*4+j] * P[((size_t)b*Ll+li)*(2*EDl) + e];
  }
  xs[i] = acc * sigm_(acc);
}
__global__ void k_ymul(const float* y, const float* P, float* o, int B, int Ll, int EDl){
  long i=(long)blockIdx.x*blockDim.x+threadIdx.x; long tot=(long)B*Ll*EDl; if(i>=tot) return;
  int e=(int)(i%EDl); int l=(int)((i/EDl)%Ll); int b=(int)(i/((long)EDl*Ll));
  float z = P[((size_t)b*Ll+l)*(2*EDl) + EDl + e];
  o[i] = y[i] * z * sigm_(z);
}
__global__ void k_scan(const float* xs, const float* delta, const float* bc,
                       const float* A_log, const float* Dp, float* y,
                       int B, int Ll, int EDl, int dtr){
  int idx = blockIdx.x*blockDim.x+threadIdx.x; if (idx >= B*EDl) return;
  int b = idx / EDl, e = idx % EDl;
  const int xw = dtr + 2*NSTN;
  float A[NSTN], h[NSTN];
#pragma unroll
  for (int n = 0; n < NSTN; ++n){ A[n] = -expf(A_log[e*NSTN+n]); h[n] = 0.f; }
  float De = Dp[e];
  for (int l = 0; l < Ll; ++l){
    size_t base = (size_t)b*Ll + l;
    float d  = delta[base*EDl + e];
    float xv = xs[base*EDl + e];
    const float* bcr = bc + base*xw + dtr;
    float acc = 0.f, dx = d * xv;
#pragma unroll
    for (int n = 0; n < NSTN; ++n){
      h[n] = expf(d*A[n])*h[n] + dx*bcr[n];
      acc += h[n]*bcr[NSTN+n];
    }
    y[base*EDl + e] = acc + De*xv;
  }
}

// ---------------- attention (per (n,head) block, S<=64 threads) ----------------
__global__ void k_attn(const float* qkv, float* out, int S, int Nn, int E, int nh){
  int n = blockIdx.x / nh, hh = blockIdx.x % nh;
  int hd = E / nh;
  int s = threadIdx.x;
  extern __shared__ float sm[];
  float* ksm = sm; float* vsm = sm + S*hd;
  const float* base = qkv + ((size_t)s*Nn + n)*3*E + hh*hd;
  for (int d = 0; d < hd; ++d){ ksm[s*hd+d] = base[E+d]; vsm[s*hd+d] = base[2*E+d]; }
  __syncthreads();
  float q[32], o[32], sc[64];
  for (int d = 0; d < hd; ++d){ q[d] = base[d]; o[d] = 0.f; }
  float scale = rsqrtf((float)hd), mx = -1e30f;
  for (int t = 0; t < S; ++t){
    float dd = 0.f;
    for (int d = 0; d < hd; ++d) dd += q[d]*ksm[t*hd+d];
    sc[t] = dd*scale; mx = fmaxf(mx, sc[t]);
  }
  float den = 0.f;
  for (int t = 0; t < S; ++t){
    float p = expf(sc[t]-mx); den += p;
    for (int d = 0; d < hd; ++d) o[d] += p*vsm[t*hd+d];
  }
  float* orow = out + ((size_t)s*Nn + n)*E + hh*hd;
  float rden = 1.f/den;
  for (int d = 0; d < hd; ++d) orow[d] = o[d]*rden;
}

// ---------------- UNet helpers ----------------
__global__ void k_maxpool(const float* X, float* Y, int B, int C, int H, int Wd){
  int Ho=H/2, Wo=Wd/2;
  long i=(long)blockIdx.x*blockDim.x+threadIdx.x; long tot=(long)B*C*Ho*Wo; if(i>=tot) return;
  int xo=(int)(i%Wo); int yo=(int)((i/Wo)%Ho); int c=(int)((i/((long)Wo*Ho))%C); int b=(int)(i/((long)Wo*Ho*C));
  const float* base = X + ((size_t)b*C + c)*H*Wd;
  float m = base[(2*yo)*Wd+2*xo];
  m = fmaxf(m, base[(2*yo)*Wd+2*xo+1]);
  m = fmaxf(m, base[(2*yo+1)*Wd+2*xo]);
  m = fmaxf(m, base[(2*yo+1)*Wd+2*xo+1]);
  Y[i]=m;
}
__global__ void k_upsample(const float* X, float* Y, int B, int C, int H, int Wd){
  int Ho=2*H, Wo=2*Wd;
  long i=(long)blockIdx.x*blockDim.x+threadIdx.x; long tot=(long)B*C*Ho*Wo; if(i>=tot) return;
  int xo=(int)(i%Wo); int yo=(int)((i/Wo)%Ho); int c=(int)((i/((long)Wo*Ho))%C); int b=(int)(i/((long)Wo*Ho*C));
  float fy = yo * (float)(H-1)/(float)(Ho-1);
  float fx = xo * (float)(Wd-1)/(float)(Wo-1);
  int ly=(int)floorf(fy), lx=(int)floorf(fx);
  int hy=min(ly+1,H-1), hx=min(lx+1,Wd-1);
  float wy=fy-ly, wx=fx-lx;
  const float* base = X + ((size_t)b*C + c)*H*Wd;
  float v = (1.f-wy)*((1.f-wx)*base[ly*Wd+lx] + wx*base[ly*Wd+hx])
          +       wy*((1.f-wx)*base[hy*Wd+lx] + wx*base[hy*Wd+hx]);
  Y[i]=v;
}
__global__ void k_pad(const float* X, float* Y, int B,int C,int Hs,int Wsd,int Hd,int Wdd,int top,int left){
  long i=(long)blockIdx.x*blockDim.x+threadIdx.x; long tot=(long)B*C*Hd*Wdd; if(i>=tot) return;
  int xo=(int)(i%Wdd); int yo=(int)((i/Wdd)%Hd); long bc=i/((long)Wdd*Hd);
  int ys=yo-top, xs=xo-left;
  Y[i] = (ys>=0 && ys<Hs && xs>=0 && xs<Wsd) ? X[bc*(size_t)Hs*Wsd + (size_t)ys*Wsd + xs] : 0.f;
}
__global__ void k_cpch(float* D, const float* S_, int B,int Cs,int Ct,int coff,int HWl){
  long i=(long)blockIdx.x*blockDim.x+threadIdx.x; long tot=(long)B*Cs*HWl; if(i>=tot) return;
  int p=(int)(i%HWl); int c=(int)((i/HWl)%Cs); int b=(int)(i/((long)HWl*Cs));
  D[((size_t)b*Ct + coff + c)*HWl + p] = S_[i];
}
__global__ void k_meanhw(const float* X, float* Y, int B,int C,int HWl){
  int i = blockIdx.x*blockDim.x+threadIdx.x; if (i>=B*C) return;
  float a=0.f; const float* p = X + (size_t)i*HWl;
  for (int j=0;j<HWl;++j) a+=p[j];
  Y[i]=a/HWl;
}
__global__ void k_getcenter(const float* X, float* V, int BC, int HWl, int pos){
  int i = blockIdx.x*blockDim.x+threadIdx.x; if (i>=BC) return;
  V[i] = X[(size_t)i*HWl + pos];
}
__global__ void k_setcenter(float* X, const float* V, int BC, int HWl, int pos){
  int i = blockIdx.x*blockDim.x+threadIdx.x; if (i>=BC) return;
  X[(size_t)i*HWl + pos] = V[i];
}

// ==================== host orchestration ====================
extern "C" void kernel_launch(void* const* d_in, const int* in_sizes, int n_in,
                              void* d_out, int out_size, void* d_ws, size_t ws_size,
                              hipStream_t stream) {
  (void)in_sizes; (void)n_in; (void)out_size; (void)ws_size;
  const float* X0 = (const float*)d_in[0];
  auto P = [&](int i){ return (const float*)d_in[i]; };

  float* ws = (float*)d_ws;
  size_t off = 0;
  auto alloc = [&](size_t n)->float*{ size_t o=off; off += (n + 63) & ~(size_t)63; return ws + o; };

  float* stats  = alloc(2048);
  float* featb  = alloc(64*256);
  float* featln = alloc(64*256);
  float* cseq   = alloc(64*ICN);
  float* cseqo  = alloc(64*ICN);
  float* xp45   = alloc((size_t)M1N*45);
  float* slotA = alloc((size_t)M1N*ICN);
  float* slotB = alloc((size_t)M1N*ICN);
  float* slotC = alloc((size_t)M1N*ICN);
  float* slotD = alloc((size_t)M1N*ICN);
  float* S1 = alloc((size_t)M1N*800);
  float* S2 = alloc((size_t)M1N*FFN);
  float* S4 = alloc((size_t)M1N*EDN);
  float* S5 = alloc((size_t)M1N*EDN);
  float* S6 = alloc((size_t)M1N*EDN);
  float* S7 = alloc((size_t)BATCHN*256*HWN);
  float* S8 = alloc((size_t)BATCHN*512*36);

  auto ew = [&](long n){ return dim3((unsigned)((n + 255) / 256)); };
  auto gemm = [&](const float* A,int lda,const float* W,const float* bias,const float* res,
                  float* C,int M,int K,int N,int act){
    dim3 g((N+31)/32, (M+63)/64);
    k_gemm_wmma<<<g,128,0,stream>>>(A,lda,W,bias,res,C,M,K,N,act);
  };
  auto conv = [&](const float* Xc,const float* Wc,const float* bc,float* Yc,
                  int Ci,int H,int Wd,int Co,int ks,int pad){
    int M = BATCHN*H*Wd;
    dim3 g((Co+31)/32, (M+63)/64);
    if (ks == 3) k_conv_wmma<3><<<g,128,0,stream>>>(Xc,Wc,bc,Yc,BATCHN,Ci,H,Wd,Co,pad);
    else         k_conv_wmma<1><<<g,128,0,stream>>>(Xc,Wc,bc,Yc,BATCHN,Ci,H,Wd,Co,pad);
  };
  auto bn = [&](float* Xb,int C,int HWl,const float* g_,const float* b_,int relu){
    k_bnstats<<<C,256,0,stream>>>(Xb,stats,BATCHN,C,HWl);
    k_bnapply<<<ew((long)BATCHN*C*HWl),256,0,stream>>>(Xb,stats,g_,b_,BATCHN,C,HWl,relu);
  };
  auto encoder = [&](const float* Xe, float* Oe, int S, int Nn, int E, int pb){
    int Me = S*Nn;
    gemm(Xe,E,P(pb+0),P(pb+1),nullptr,S1,Me,E,3*E,0);
    int hd = E/NHEADN;
    k_attn<<<Nn*NHEADN,S,(size_t)2*S*hd*sizeof(float),stream>>>(S1,S5,S,Nn,E,NHEADN);
    gemm(S5,E,P(pb+2),P(pb+3),Xe,S6,Me,E,E,0);
    k_layernorm<<<Me,128,0,stream>>>(S6,P(pb+4),P(pb+5),S4,Me,E);
    gemm(S4,E,P(pb+6),P(pb+7),nullptr,S2,Me,E,FFN,1);
    gemm(S2,FFN,P(pb+8),P(pb+9),S4,S6,Me,FFN,E,0);
    k_layernorm<<<Me,128,0,stream>>>(S6,P(pb+10),P(pb+11),S4,Me,E);
    k_layernorm<<<Me,128,0,stream>>>(S4,P(pb+12),P(pb+13),Oe,Me,E);
  };

  const long NX = (long)M1N*ICN;
  const int CPOS = 6*WSN + 6;

  // ---- Mamba block (params 1..10); mamba2 (11..20) is dead code in the reference ----
  k_nchw2seq<<<ew(NX),256,0,stream>>>(slotA, X0, BATCHN, ICN, HWN);
  k_rmsnorm<<<M1N,128,0,stream>>>(slotA, P(1), slotB, M1N, ICN);
  gemm(slotB,ICN, P(2), nullptr, nullptr, S1, M1N, ICN, 2*EDN, 0);
  k_dwconv_silu<<<ew((long)M1N*EDN),256,0,stream>>>(S1, P(3), P(4), S4, BATCHN, HWN, EDN);
  gemm(S4,EDN, P(5), nullptr, nullptr, xp45, M1N, EDN, DTRN+2*NSTN, 0);
  gemm(xp45,DTRN+2*NSTN, P(6), P(7), nullptr, S5, M1N, DTRN, EDN, 2);
  k_scan<<<ew((long)BATCHN*EDN),256,0,stream>>>(S4,S5,xp45,P(8),P(9),S6,BATCHN,HWN,EDN,DTRN);
  k_ymul<<<ew((long)M1N*EDN),256,0,stream>>>(S6, S1, S4, BATCHN, HWN, EDN);
  gemm(S4,EDN, P(10), nullptr, slotA, slotB, M1N, EDN, ICN, 0);
  k_seq2nchw<<<ew(NX),256,0,stream>>>(slotC, slotB, BATCHN, ICN, HWN);

  // ---- BatchNorm2d ----
  bn(slotC, ICN, HWN, P(21), P(22), 0);

  // ---- channel_attn on center pixel (enc0) ----
  k_getcenter<<<ew(BATCHN*ICN),256,0,stream>>>(slotC, cseq, BATCHN*ICN, HWN, CPOS);
  encoder(cseq, cseqo, BATCHN, 1, ICN, 23);
  k_copy<<<ew(NX),256,0,stream>>>(slotD, slotC, NX);
  k_setcenter<<<ew(BATCHN*ICN),256,0,stream>>>(slotD, cseqo, BATCHN*ICN, HWN, CPOS);

  // ---- + pos_emb, encoder_layers (enc1), + x1 ----
  k_addip<<<ew(NX),256,0,stream>>>(slotC, P(51), NX);
  k_nchw2seq<<<ew(NX),256,0,stream>>>(slotB, slotC, BATCHN, ICN, HWN);
  encoder(slotB, slotA, BATCHN, HWN, ICN, 37);
  k_seq2nchw<<<ew(NX),256,0,stream>>>(slotC, slotA, BATCHN, ICN, HWN);
  k_addip<<<ew(NX),256,0,stream>>>(slotC, slotD, NX);

  // ---- UNet encoder ----
  conv(slotC, P(52), P(53), S1, ICN, 13,13, 256, 3,1);  bn(S1, 256,169, P(54),P(55),1);
  conv(S1,   P(56), P(57), S7, 256, 13,13, 256, 3,1);   bn(S7, 256,169, P(58),P(59),1);
  k_maxpool<<<ew((long)BATCHN*256*36),256,0,stream>>>(S7, slotA, BATCHN,256,13,13);
  conv(slotA,P(60), P(61), slotB, 256, 6,6, 512, 3,1);  bn(slotB,512,36, P(62),P(63),1);
  conv(slotB,P(64), P(65), S8,   512, 6,6, 512, 3,1);   bn(S8,  512,36, P(66),P(67),1);
  k_maxpool<<<ew((long)BATCHN*512*9),256,0,stream>>>(S8, slotA, BATCHN,512,6,6);
  conv(slotA,P(68), P(69), slotB, 512, 3,3, 1024,3,1);  bn(slotB,1024,9, P(70),P(71),1);
  conv(slotB,P(72), P(73), slotD, 1024,3,3, 1024,3,1);  bn(slotD,1024,9, P(74),P(75),1);

  // ---- up1 ----
  k_upsample<<<ew((long)BATCHN*1024*36),256,0,stream>>>(slotD, S4, BATCHN,1024,3,3);
  conv(S4, P(76), P(77), S5, 1024, 6,6, 512, 1,0);
  k_cpch<<<ew((long)BATCHN*512*36),256,0,stream>>>(S6, S8, BATCHN,512,1024,0,36);
  k_cpch<<<ew((long)BATCHN*512*36),256,0,stream>>>(S6, S5, BATCHN,512,1024,512,36);
  conv(S6, P(78), P(79), S2, 1024, 6,6, 512, 3,1);      bn(S2, 512,36, P(80),P(81),1);
  conv(S2, P(82), P(83), S4, 512,  6,6, 512, 3,1);      bn(S4, 512,36, P(84),P(85),1);

  // ---- up2 ----
  k_upsample<<<ew((long)BATCHN*512*144),256,0,stream>>>(S4, S1, BATCHN,512,6,6);
  conv(S1, P(86), P(87), S5, 512, 12,12, 256, 1,0);
  k_pad<<<ew((long)BATCHN*256*169),256,0,stream>>>(S5, S6, BATCHN,256,12,12,13,13,0,0);
  k_cpch<<<ew((long)BATCHN*256*169),256,0,stream>>>(S2, S7, BATCHN,256,512,0,169);
  k_cpch<<<ew((long)BATCHN*256*169),256,0,stream>>>(S2, S6, BATCHN,256,512,256,169);
  conv(S2, P(88), P(89), S1, 512, 13,13, 256, 3,1);     bn(S1, 256,169, P(90),P(91),1);
  conv(S1, P(92), P(93), S6, 256, 13,13, 256, 3,1);     bn(S6, 256,169, P(94),P(95),1);

  // ---- head ----
  k_meanhw<<<ew(BATCHN*256),256,0,stream>>>(S6, featb, BATCHN, 256, 169);
  k_layernorm<<<BATCHN,128,0,stream>>>(featb, P(96), P(97), featln, BATCHN, 256);
  gemm(featln, 256, P(98), P(99), nullptr, (float*)d_out, BATCHN, 256, 16, 0);
}
